// Graphformer_Block_19859928777408
// MI455X (gfx1250) — matrix operand order repaired and checked by reading, not compiled, not run on hardware
//
#include <hip/hip_runtime.h>
#include <math.h>

// ---------------------------------------------------------------------------
// Graphformer block for MI455X (gfx1250, wave32, WMMA + TDM).
// All GEMM-shaped stages use v_wmma_f32_16x16x32_f16 (f16 A/B, f32 acc).
// QKV GEMM stages its strided B-panel into LDS via the Tensor Data Mover
// (tensor_load_to_lds + s_wait_tensorcnt). BN + residual + leaky fused into
// GEMM epilogues. EMA blocks keep their group tile resident in LDS.
// ---------------------------------------------------------------------------

typedef __attribute__((ext_vector_type(16))) _Float16 v16h;
typedef __attribute__((ext_vector_type(8)))  float    v8f;
typedef unsigned int u32x4 __attribute__((ext_vector_type(4)));
typedef int          i32x4 __attribute__((ext_vector_type(4)));
typedef int          i32x8 __attribute__((ext_vector_type(8)));

#define EPSV 1e-5f

#define NB  64
#define CC  64
#define TT  120
#define VV  25
#define HH  3
#define DD  64
#define TV  3000      // T*V
#define CT  7680      // C*T (also D*T attention K-depth)

// -------------------------- WMMA helpers -----------------------------------

__device__ __forceinline__ v8f wmma_fp16(v16h a, v16h b, v8f c) {
  // 8 args: (neg_a, A, neg_b, B, c_mod, C, reuse_a, reuse_b)
  return __builtin_amdgcn_wmma_f32_16x16x32_f16(false, a, false, b, (short)0, c,
                                                false, false);
}

// A fragment: 16x32 (MxK). lanes 0-15: m=lane, K-half 0; lanes 16-31: K-half 1.
template <typename F>
__device__ __forceinline__ v16h load_a_frag(F f, int lane) {
  int m  = lane & 15;
  int kh = (lane >> 4) * 8;
  v16h a;
#pragma unroll
  for (int j = 0; j < 4; ++j) {
    a[2 * j]     = f(m, kh + 2 * j);
    a[2 * j + 1] = f(m, kh + 2 * j + 1);
    a[2 * j + 8] = f(m, 16 + kh + 2 * j);
    a[2 * j + 9] = f(m, 16 + kh + 2 * j + 1);
  }
  return a;
}

// B fragment: 32x16 (KxN). lane holds column n=lane&15; lanes 0-15: K=0..7,16..23,
// lanes 16-31: K=8..15,24..31 (same k order as the A fragment).
template <typename F>
__device__ __forceinline__ v16h load_b_frag(F f, int lane) {
  int n  = lane & 15;
  int kb = (lane >> 4) * 8;
  v16h b;
#pragma unroll
  for (int j = 0; j < 4; ++j) {
    b[2 * j]     = f(kb + 2 * j, n);
    b[2 * j + 1] = f(kb + 2 * j + 1, n);
    b[2 * j + 8] = f(16 + kb + 2 * j, n);
    b[2 * j + 9] = f(16 + kb + 2 * j + 1, n);
  }
  return b;
}

__device__ __forceinline__ float leakyf(float x) { return x >= 0.f ? x : 0.1f * x; }
__device__ __forceinline__ float sigmoidf(float x) { return 1.f / (1.f + expf(-x)); }

// -------------------------- prep kernels -----------------------------------

// xs = x + pos_embed  (pe depends only on (c,v)); also keep plain x as f16.
__global__ __launch_bounds__(256)
void prep_x_kernel(const float* __restrict__ x,
                   _Float16* __restrict__ XSF,   // x + pe, f16
                   _Float16* __restrict__ XF) {  // x, f16
  size_t i = (size_t)blockIdx.x * 256 + threadIdx.x;
  if (i >= (size_t)NB * CC * TV) return;
  int v = (int)(i % VV);
  int c = (int)((i / TV) % CC);
  float xv = x[i];
  int half = c >> 1;
  float div = expf((float)(2 * half) * (-9.210340371976184f / (float)CC));
  float pe  = (c & 1) ? cosf((float)v * div) : sinf((float)v * div);
  XSF[i] = (_Float16)(xv + pe);
  XF[i]  = (_Float16)xv;
}

// Convert qkv_w to f16; repack conv weights into (K x 64) B-matrices.
__global__ __launch_bounds__(256)
void prep_w_kernel(const float* __restrict__ qkv_w,   // (384,64)
                   const float* __restrict__ nets_w,  // (64,192,1,3)
                   const float* __restrict__ nett_w,  // (64,64,5,1)
                   _Float16* __restrict__ WQ,         // 384x64
                   _Float16* __restrict__ WS,         // 576x64: K=(ci*3+r)
                   _Float16* __restrict__ WT) {       // 320x64: K=(ci*5+r)
  int i = blockIdx.x * 256 + threadIdx.x;
  if (i < 384 * 64) WQ[i] = (_Float16)qkv_w[i];
  if (i < 576 * 64) {
    int k = i / 64, o = i % 64;
    int ci = k / 3, r = k % 3;
    WS[i] = (_Float16)nets_w[(o * 192 + ci) * 3 + r];
  }
  if (i < 320 * 64) {
    int k = i / 64, o = i % 64;
    int ci = k / 5, r = k % 5;
    WT[i] = (_Float16)nett_w[(o * 64 + ci) * 5 + r];
  }
}

// -------------------------- GEMM kernels (WMMA) -----------------------------

// qk[n,o,p] = sum_c W[o,c] * xs[n,c,p] + qkv_b[o].  M=384, N=3000, K=64.
// B-panel (64 channels x 64 cols, the strided side) staged into LDS by the
// Tensor Data Mover; 4 waves then sweep all 24 M-tiles against the panel.
__global__ __launch_bounds__(128)
void qkv_gemm_kernel(const _Float16* __restrict__ W,   // 384x64 (f16, row-major)
                     const _Float16* __restrict__ XS,  // N x 64 x 3000
                     const float* __restrict__ qkv_b,
                     _Float16* __restrict__ QK) {      // N x 384 x 3000
  __shared__ _Float16 Bp[CC][64];  // [channel][col - c0], 8 KB
  int lane = threadIdx.x & 31, wave = threadIdx.x >> 5;
  int n  = blockIdx.x;
  int c0 = blockIdx.y * 64;        // column tile base (47 tiles >= 3000)
  const _Float16* X = XS + (size_t)n * CC * TV;

  if (wave == 0) {
    // Tensor DMA descriptor (CDNA5 ISA ch.8): 2D tile, 2-byte elements,
    // tile 64 (cols) x 64 (rows=channels), row stride = T*V elements.
    unsigned long long ga = (unsigned long long)(uintptr_t)(X + c0);
    unsigned int lds = (unsigned int)(uintptr_t)(void*)&Bp[0][0];
    u32x4 g0;
    g0.x = 1u;                                      // count=1, user descriptor
    g0.y = lds;                                     // lds_addr (bytes)
    g0.z = (unsigned int)ga;                        // global_addr[31:0]
    g0.w = (unsigned int)((ga >> 32) & 0x01FFFFFFu) // global_addr[56:32]
           | (2u << 30);                            // type = 2 ("image")
    i32x8 g1;
    g1[0] = 1 << 16;                 // workgroup_mask=0, data_size=1 (2 bytes)
    g1[1] = (TV & 0xFFFF) << 16;     // tensor_dim0[15:0]   (bits 63:48)
    g1[2] = (TV >> 16) | (CC << 16); // tensor_dim0[31:16] | tensor_dim1[15:0]
    g1[3] = (64 << 16);              // tensor_dim1[31:16]=0 | tile_dim0=64
    g1[4] = 64;                      // tile_dim1=64, tile_dim2=0
    g1[5] = TV;                      // tensor_dim0_stride[31:0]
    g1[6] = 0;                       // stride hi bits / dim1_stride (unused 2D)
    g1[7] = 0;
    i32x4 z4 = {};
#if defined(__clang_major__) && (__clang_major__ >= 23)
    i32x8 z8 = {};
    __builtin_amdgcn_tensor_load_to_lds(g0, g1, z4, z4, z8, 0);
#else
    __builtin_amdgcn_tensor_load_to_lds(g0, g1, z4, z4, 0);
#endif
    __builtin_amdgcn_s_wait_tensorcnt(0);
  }
  __syncthreads();

  int colb = wave * 16;
  _Float16* O = QK + (size_t)n * 384 * TV;
  for (int mt = 0; mt < 24; ++mt) {
    int m0 = mt * 16;
    v8f acc = {};
#pragma unroll
    for (int k0 = 0; k0 < CC; k0 += 32) {
      v16h a = load_a_frag([&](int mm, int kk) -> _Float16 {
        return W[(m0 + mm) * CC + k0 + kk];
      }, lane);
      v16h b = load_b_frag([&](int kk, int nn) -> _Float16 {
        return Bp[k0 + kk][colb + nn];
      }, lane);
      acc = wmma_fp16(a, b, acc);
    }
    int nn = lane & 15, mb = (lane >> 4) * 8;
    int col = c0 + colb + nn;
    if (col < TV) {
#pragma unroll
      for (int r = 0; r < 8; ++r) {
        int o = m0 + mb + r;
        O[(size_t)o * TV + col] = (_Float16)(acc[r] + qkv_b[o]);
      }
    }
  }
}

// att[n,h,u,v] = tanh(sum_{d,t} q*k / (D*T)) * alpha * atten_w + att0s.
// Per (n,h): M=N=25 (pad 32), K = D*T = 7680 (k index = t*64 + d).
__global__ __launch_bounds__(128)
void att_kernel(const _Float16* __restrict__ QK,
                const float* __restrict__ atten_w,  // 25x25
                const float* __restrict__ alphas,   // H
                const float* __restrict__ att0s,    // H*625
                _Float16* __restrict__ ATT) {       // (N*H) x 625
  int lane = threadIdx.x & 31, wave = threadIdx.x >> 5;
  int nh = blockIdx.x;
  int n = nh / HH, h = nh % HH;
  int mu0 = (wave >> 1) * 16, nv0 = (wave & 1) * 16;
  const _Float16* Q  = QK + (size_t)n * 384 * TV + (size_t)(h * DD) * TV;
  const _Float16* Kk = QK + (size_t)n * 384 * TV + (size_t)(HH * DD + h * DD) * TV;
  v8f acc = {};
  for (int k0 = 0; k0 < CT; k0 += 32) {
    v16h a = load_a_frag([&](int mm, int kk) -> _Float16 {
      int u = mu0 + mm; u = u < VV ? u : VV - 1;   // dup rows never stored
      int kg = k0 + kk; int d = kg & 63, t = kg >> 6;
      return Q[(size_t)d * TV + t * VV + u];
    }, lane);
    v16h b = load_b_frag([&](int kk, int nn) -> _Float16 {
      int v = nv0 + nn; v = v < VV ? v : VV - 1;
      int kg = k0 + kk; int d = kg & 63, t = kg >> 6;
      return Kk[(size_t)d * TV + t * VV + v];
    }, lane);
    acc = wmma_fp16(a, b, acc);
  }
  int nn = lane & 15, mb = (lane >> 4) * 8;
  int v = nv0 + nn;
  if (v < VV) {
#pragma unroll
    for (int r = 0; r < 8; ++r) {
      int u = mu0 + mb + r;
      if (u < VV) {
        float av = tanhf(acc[r] * (1.0f / (float)(DD * TT))) * alphas[h];
        av = av * atten_w[u * VV + v] + att0s[(h * VV + u) * VV + v];
        ATT[(size_t)nh * 625 + u * VV + v] = (_Float16)av;
      }
    }
  }
}

// xs2[n, h*64+c, t, v] = sum_u x[n,c,t,u] * att[n,h,u,v].
// Per (n,h): A = x[n] as (7680 x 25), B = att (25 x 25), K padded to 32.
__global__ __launch_bounds__(128)
void xs2_gemm_kernel(const _Float16* __restrict__ XF,   // N x 64 x 3000
                     const _Float16* __restrict__ ATT,  // (N*H) x 625
                     _Float16* __restrict__ XS2) {      // N x 192 x 3000
  int lane = threadIdx.x & 31, wave = threadIdx.x >> 5;
  int nh = blockIdx.x;
  int n = nh / HH, h = nh % HH;
  int m0 = (blockIdx.y * 4 + wave) * 16;  // row tile over C*T = 7680 (480 tiles)
  const _Float16* A = XF + (size_t)n * CC * TV;
  const _Float16* B = ATT + (size_t)nh * 625;
  v8f acc0 = {}, acc1 = {};
  v16h af = load_a_frag([&](int mm, int kk) -> _Float16 {
    return kk < VV ? A[(size_t)(m0 + mm) * VV + kk] : (_Float16)0.f;
  }, lane);
  v16h bf0 = load_b_frag([&](int kk, int nn) -> _Float16 {
    return (kk < VV && nn < VV) ? B[kk * VV + nn] : (_Float16)0.f;
  }, lane);
  v16h bf1 = load_b_frag([&](int kk, int nn) -> _Float16 {
    int v = 16 + nn;
    return (kk < VV && v < VV) ? B[kk * VV + v] : (_Float16)0.f;
  }, lane);
  acc0 = wmma_fp16(af, bf0, acc0);
  acc1 = wmma_fp16(af, bf1, acc1);

  int nn = lane & 15, mb = (lane >> 4) * 8;
  _Float16* O = XS2 + ((size_t)n * 192 + h * CC) * TV;
#pragma unroll
  for (int r = 0; r < 8; ++r) {
    int row = m0 + mb + r;              // (c,t)
    int c = row / TT, t = row % TT;
    size_t base = (size_t)c * TV + (size_t)t * VV;
    if (nn < VV)      O[base + nn]      = (_Float16)acc0[r];
    if (16 + nn < VV) O[base + 16 + nn] = (_Float16)acc1[r];
  }
}

// conv 1x3 over V (192 in-ch) as im2col GEMM: K = 192*3 = 576.
// Epilogue: BN1 + residual x + leaky -> S1 (f32).
__global__ __launch_bounds__(128)
void convS_kernel(const _Float16* __restrict__ XS2,  // N x 192 x 3000
                  const _Float16* __restrict__ WS,   // 576 x 64
                  const float* __restrict__ bias, const float* __restrict__ g,
                  const float* __restrict__ bb, const float* __restrict__ bm,
                  const float* __restrict__ bv, const float* __restrict__ RES,
                  float* __restrict__ S1) {
  int lane = threadIdx.x & 31, wave = threadIdx.x >> 5;
  int n  = blockIdx.x;
  int m0 = blockIdx.y * 16;   // position tile (188)
  int n0 = wave * 16;         // output-channel tile
  const _Float16* Xn = XS2 + (size_t)n * 192 * TV;
  v8f acc = {};
  for (int k0 = 0; k0 < 576; k0 += 32) {
    v16h a = load_a_frag([&](int mm, int kk) -> _Float16 {
      int p = m0 + mm; p = p < TV ? p : TV - 1;
      int t = p / VV, v = p % VV;
      int kg = k0 + kk; int ci = kg / 3, r = kg % 3;
      int v2 = v + r - 1;
      return (v2 >= 0 && v2 < VV) ? Xn[(size_t)ci * TV + t * VV + v2]
                                  : (_Float16)0.f;
    }, lane);
    v16h bf = load_b_frag([&](int kk, int nn) -> _Float16 {
      return WS[(size_t)(k0 + kk) * 64 + n0 + nn];
    }, lane);
    acc = wmma_fp16(a, bf, acc);
  }
  int nn = lane & 15, mb = (lane >> 4) * 8;
  int o = n0 + nn;
  float sc = g[o] / sqrtf(bv[o] + EPSV);
#pragma unroll
  for (int r = 0; r < 8; ++r) {
    int p = m0 + mb + r;
    if (p < TV) {
      size_t idx = ((size_t)n * CC + o) * TV + p;
      float y = (acc[r] + bias[o] - bm[o]) * sc + bb[o] + RES[idx];
      S1[idx] = leakyf(y);
    }
  }
}

// conv 5x1 over T as im2col GEMM: K = 64*5 = 320.
// Epilogue: BN3 + residual s2 + leaky -> S3 (f32).
__global__ __launch_bounds__(128)
void convT_kernel(const _Float16* __restrict__ S2H,  // N x 64 x 3000
                  const _Float16* __restrict__ WT,   // 320 x 64
                  const float* __restrict__ bias, const float* __restrict__ g,
                  const float* __restrict__ bb, const float* __restrict__ bm,
                  const float* __restrict__ bv, const float* __restrict__ RES,
                  float* __restrict__ S3) {
  int lane = threadIdx.x & 31, wave = threadIdx.x >> 5;
  int n  = blockIdx.x;
  int m0 = blockIdx.y * 16;
  int n0 = wave * 16;
  const _Float16* Xn = S2H + (size_t)n * CC * TV;
  v8f acc = {};
  for (int k0 = 0; k0 < 320; k0 += 32) {
    v16h a = load_a_frag([&](int mm, int kk) -> _Float16 {
      int p = m0 + mm; p = p < TV ? p : TV - 1;
      int t = p / VV, v = p % VV;
      int kg = k0 + kk; int ci = kg / 5, r = kg % 5;
      int t2 = t + r - 2;
      return (t2 >= 0 && t2 < TT) ? Xn[(size_t)ci * TV + t2 * VV + v]
                                  : (_Float16)0.f;
    }, lane);
    v16h bf = load_b_frag([&](int kk, int nn) -> _Float16 {
      return WT[(size_t)(k0 + kk) * 64 + n0 + nn];
    }, lane);
    acc = wmma_fp16(a, bf, acc);
  }
  int nn = lane & 15, mb = (lane >> 4) * 8;
  int o = n0 + nn;
  float sc = g[o] / sqrtf(bv[o] + EPSV);
#pragma unroll
  for (int r = 0; r < 8; ++r) {
    int p = m0 + mb + r;
    if (p < TV) {
      size_t idx = ((size_t)n * CC + o) * TV + p;
      float y = (acc[r] + bias[o] - bm[o]) * sc + bb[o] + RES[idx];
      S3[idx] = leakyf(y);
    }
  }
}

// -------------------------- EMA kernels -------------------------------------

__device__ __forceinline__ float block_reduce_sum(float v, float* red, int tid) {
  red[tid] = v;
  __syncthreads();
  for (int s = 128; s > 0; s >>= 1) {
    if (tid < s) red[tid] += red[tid + s];
    __syncthreads();
  }
  float r = red[0];
  __syncthreads();
  return r;
}

// One workgroup per group-image (cg channels x 120 x 25 resident in LDS).
// Fused epilogue: BN + residual + leaky; optional f16 copy for the next GEMM.
template <int CG>
__global__ __launch_bounds__(256)
void ema_kernel(const float* __restrict__ in, const float* __restrict__ res,
                const float* __restrict__ c1w, const float* __restrict__ c1b,
                const float* __restrict__ c3w, const float* __restrict__ c3b,
                const float* __restrict__ gng, const float* __restrict__ gnb,
                const float* __restrict__ bng, const float* __restrict__ bnb,
                const float* __restrict__ bnm, const float* __restrict__ bnv,
                float* __restrict__ out32, _Float16* __restrict__ out16) {
  constexpr int GROUPS = CC / CG;
  __shared__ float gx[CG][TV];
  __shared__ float xh[CG][TT], xw[CG][VV];
  __shared__ float sh[CG][TT], sw[CG][VV];
  __shared__ float red[256];
  __shared__ float mu[CG], rstd[CG], x2m[CG], x11[CG], x21[CG];

  const int tid = threadIdx.x;
  const int bg = blockIdx.x;
  const int b = bg / GROUPS, gidx = bg % GROUPS;
  const int cbase = gidx * CG;
  const float* src = in + ((size_t)b * CC + cbase) * TV;

  for (int i = tid; i < CG * TV; i += 256) gx[i / TV][i % TV] = src[i];
  __syncthreads();

  for (int i = tid; i < CG * TT; i += 256) {
    int c = i / TT, t = i % TT;
    float s = 0.f;
    for (int v = 0; v < VV; ++v) s += gx[c][t * VV + v];
    xh[c][t] = s * (1.f / VV);
  }
  for (int i = tid; i < CG * VV; i += 256) {
    int c = i / VV, v = i % VV;
    float s = 0.f;
    for (int t = 0; t < TT; ++t) s += gx[c][t * VV + v];
    xw[c][v] = s * (1.f / TT);
  }
  __syncthreads();

  // 1x1 conv over concat([x_h; x_w]) + sigmoid gates.
  for (int i = tid; i < CG * (TT + VV); i += 256) {
    int c2 = i / (TT + VV), j = i % (TT + VV);
    float s = c1b[c2];
    for (int c1 = 0; c1 < CG; ++c1)
      s += c1w[c2 * CG + c1] * (j < TT ? xh[c1][j] : xw[c1][j - TT]);
    float sg = sigmoidf(s);
    if (j < TT) sh[c2][j] = sg; else sw[c2][j - TT] = sg;
  }
  __syncthreads();

  // Pass 1: per-channel mean/var of gated g, and 3x3-conv channel means.
  float a0[CG], a1[CG], a2[CG];
#pragma unroll
  for (int c = 0; c < CG; ++c) a0[c] = a1[c] = a2[c] = 0.f;
  for (int p = tid; p < TV; p += 256) {
    int t = p / VV, v = p % VV;
#pragma unroll
    for (int c = 0; c < CG; ++c) {
      float gv = gx[c][p] * sh[c][t] * sw[c][v];
      a0[c] += gv; a1[c] += gv * gv;
      float s = 0.f;
      for (int c1 = 0; c1 < CG; ++c1)
        for (int dt = -1; dt <= 1; ++dt) {
          int t2 = t + dt; if (t2 < 0 || t2 >= TT) continue;
          for (int dv = -1; dv <= 1; ++dv) {
            int v2 = v + dv; if (v2 < 0 || v2 >= VV) continue;
            s += c3w[((c * CG + c1) * 3 + dt + 1) * 3 + dv + 1] *
                 gx[c1][t2 * VV + v2];
          }
        }
      a2[c] += s;
    }
  }
#pragma unroll
  for (int c = 0; c < CG; ++c) {
    float s0 = block_reduce_sum(a0[c], red, tid);
    float s1 = block_reduce_sum(a1[c], red, tid);
    float s2 = block_reduce_sum(a2[c], red, tid);
    if (tid == 0) {
      float m_ = s0 * (1.f / TV);
      mu[c] = m_;
      rstd[c] = 1.f / sqrtf(s1 * (1.f / TV) - m_ * m_ + EPSV);
      x2m[c] = s2 * (1.f / TV) + c3b[c];
    }
  }
  __syncthreads();
  if (tid == 0) {
    // x1.mean((2,3)) == gnb exactly (groupnorm output channel mean).
    float m1 = gnb[0], m2 = x2m[0];
    for (int c = 1; c < CG; ++c) { m1 = fmaxf(m1, gnb[c]); m2 = fmaxf(m2, x2m[c]); }
    float d1 = 0.f, d2 = 0.f;
    for (int c = 0; c < CG; ++c) {
      x11[c] = expf(gnb[c] - m1); d1 += x11[c];
      x21[c] = expf(x2m[c] - m2); d2 += x21[c];
    }
    for (int c = 0; c < CG; ++c) { x11[c] /= d1; x21[c] /= d2; }
  }
  __syncthreads();

  // Pass 2: weights = x11·x2 + x21·x1 ; out = gx * sigmoid(weights); BN+res+leaky.
  for (int p = tid; p < TV; p += 256) {
    int t = p / VV, v = p % VV;
    float wsum = 0.f;
#pragma unroll
    for (int c = 0; c < CG; ++c) {
      float gv = gx[c][p] * sh[c][t] * sw[c][v];
      float x1v = (gv - mu[c]) * rstd[c] * gng[c] + gnb[c];
      float x2v = c3b[c];
      for (int c1 = 0; c1 < CG; ++c1)
        for (int dt = -1; dt <= 1; ++dt) {
          int t2 = t + dt; if (t2 < 0 || t2 >= TT) continue;
          for (int dv = -1; dv <= 1; ++dv) {
            int v2 = v + dv; if (v2 < 0 || v2 >= VV) continue;
            x2v += c3w[((c * CG + c1) * 3 + dt + 1) * 3 + dv + 1] *
                   gx[c1][t2 * VV + v2];
          }
        }
      wsum += x11[c] * x2v + x21[c] * x1v;
    }
    float sg = sigmoidf(wsum);
#pragma unroll
    for (int c = 0; c < CG; ++c) {
      int ch = cbase + c;
      size_t idx = ((size_t)b * CC + ch) * TV + p;
      float o = gx[c][p] * sg;
      float sc = bng[ch] / sqrtf(bnv[ch] + EPSV);
      float y = (o - bnm[ch]) * sc + bnb[ch] + res[idx];
      y = leakyf(y);
      out32[idx] = y;
      if (out16) out16[idx] = (_Float16)y;
    }
  }
}

// -------------------------- workspace layout --------------------------------
static constexpr size_t NCTV_   = (size_t)NB * CC * TV;          // 12,288,000
static constexpr size_t OFF_XSF = 0;                              // x+pe f16
static constexpr size_t OFF_XF  = OFF_XSF + NCTV_ * 2;            // x f16
static constexpr size_t OFF_WQ  = OFF_XF + NCTV_ * 2;             // 384x64 f16
static constexpr size_t OFF_WS  = OFF_WQ + (size_t)384 * 64 * 2;  // 576x64 f16
static constexpr size_t OFF_WT  = OFF_WS + (size_t)576 * 64 * 2;  // 320x64 f16
static constexpr size_t OFF_ATT = OFF_WT + (size_t)320 * 64 * 2;  // (N*H)x625 f16
static constexpr size_t OFF_QK  =
    (OFF_ATT + (size_t)NB * HH * 625 * 2 + 255) & ~(size_t)255;   // N*384*TV f16
static constexpr size_t SZ_QK   = (size_t)NB * 384 * TV * 2;      // reused as XS2
static constexpr size_t OFF_S1  = OFF_QK + SZ_QK;                 // f32; reused as S3
static constexpr size_t OFF_S2  = OFF_S1 + NCTV_ * 4;             // f32
static constexpr size_t OFF_S2H = OFF_S2 + NCTV_ * 4;             // f16
// total ≈ 320 MB

// -------------------------- launch ------------------------------------------
extern "C" void kernel_launch(void* const* d_in, const int* in_sizes, int n_in,
                              void* d_out, int out_size, void* d_ws, size_t ws_size,
                              hipStream_t stream) {
  (void)in_sizes; (void)n_in; (void)out_size; (void)ws_size;
  const float* x       = (const float*)d_in[0];
  const float* atten_w = (const float*)d_in[1];
  const float* qkv_w   = (const float*)d_in[2];
  const float* qkv_b   = (const float*)d_in[3];
  const float* alphas  = (const float*)d_in[4];
  const float* att0s   = (const float*)d_in[5];
  const float* nets_w  = (const float*)d_in[6];
  const float* nets_b  = (const float*)d_in[7];
  const float* bn1_g = (const float*)d_in[8];
  const float* bn1_b = (const float*)d_in[9];
  const float* bn1_m = (const float*)d_in[10];
  const float* bn1_v = (const float*)d_in[11];
  const float* e1_c1w = (const float*)d_in[12];
  const float* e1_c1b = (const float*)d_in[13];
  const float* e1_c3w = (const float*)d_in[14];
  const float* e1_c3b = (const float*)d_in[15];
  const float* e1_gng = (const float*)d_in[16];
  const float* e1_gnb = (const float*)d_in[17];
  const float* bn2_g = (const float*)d_in[18];
  const float* bn2_b = (const float*)d_in[19];
  const float* bn2_m = (const float*)d_in[20];
  const float* bn2_v = (const float*)d_in[21];
  const float* nett_w = (const float*)d_in[22];
  const float* nett_b = (const float*)d_in[23];
  const float* bn3_g = (const float*)d_in[24];
  const float* bn3_b = (const float*)d_in[25];
  const float* bn3_m = (const float*)d_in[26];
  const float* bn3_v = (const float*)d_in[27];
  const float* e2_c1w = (const float*)d_in[28];
  const float* e2_c1b = (const float*)d_in[29];
  const float* e2_c3w = (const float*)d_in[30];
  const float* e2_c3b = (const float*)d_in[31];
  const float* e2_gng = (const float*)d_in[32];
  const float* e2_gnb = (const float*)d_in[33];
  const float* bn4_g = (const float*)d_in[34];
  const float* bn4_b = (const float*)d_in[35];
  const float* bn4_m = (const float*)d_in[36];
  const float* bn4_v = (const float*)d_in[37];

  char* ws = (char*)d_ws;
  _Float16* XSF = (_Float16*)(ws + OFF_XSF);
  _Float16* XF  = (_Float16*)(ws + OFF_XF);
  _Float16* WQ  = (_Float16*)(ws + OFF_WQ);
  _Float16* WSm = (_Float16*)(ws + OFF_WS);
  _Float16* WTm = (_Float16*)(ws + OFF_WT);
  _Float16* ATT = (_Float16*)(ws + OFF_ATT);
  _Float16* QK  = (_Float16*)(ws + OFF_QK);
  _Float16* XS2 = (_Float16*)(ws + OFF_QK);  // reuse (QK dead after att)
  float*    S1  = (float*)(ws + OFF_S1);
  float*    S3  = (float*)(ws + OFF_S1);     // reuse (S1 dead after ema1)
  float*    S2  = (float*)(ws + OFF_S2);
  _Float16* S2H = (_Float16*)(ws + OFF_S2H);

  prep_x_kernel<<<48000, 256, 0, stream>>>(x, XSF, XF);
  prep_w_kernel<<<144, 256, 0, stream>>>(qkv_w, nets_w, nett_w, WQ, WSm, WTm);
  qkv_gemm_kernel<<<dim3(NB, 47), 128, 0, stream>>>(WQ, XSF, qkv_b, QK);
  att_kernel<<<NB * HH, 128, 0, stream>>>(QK, atten_w, alphas, att0s, ATT);
  xs2_gemm_kernel<<<dim3(NB * HH, 120), 128, 0, stream>>>(XF, ATT, XS2);
  convS_kernel<<<dim3(NB, 188), 128, 0, stream>>>(XS2, WSm, nets_b, bn1_g, bn1_b,
                                                  bn1_m, bn1_v, x, S1);
  ema_kernel<4><<<NB * 16, 256, 0, stream>>>(S1, x, e1_c1w, e1_c1b, e1_c3w,
                                             e1_c3b, e1_gng, e1_gnb, bn2_g,
                                             bn2_b, bn2_m, bn2_v, S2, S2H);
  convT_kernel<<<dim3(NB, 188), 128, 0, stream>>>(S2H, WTm, nett_b, bn3_g, bn3_b,
                                                  bn3_m, bn3_v, S2, S3);
  ema_kernel<2><<<NB * 32, 256, 0, stream>>>(S3, S2, e2_c1w, e2_c1b, e2_c3w,
                                             e2_c3b, e2_gng, e2_gnb, bn4_g,
                                             bn4_b, bn4_m, bn4_v, (float*)d_out,
                                             nullptr);
}
